// FPModule_62895501082990
// MI455X (gfx1250) — matrix operand order repaired and checked by reading, not compiled, not run on hardware
//
#include <hip/hip_runtime.h>

#define NC    1024   // coarse points per cloud
#define NF    4096   // fine points per cloud
#define NB    16     // batches
#define CIN   256
#define CSK   128
#define CH1   384    // CIN + CSK (GEMM1 K)
#define CMID  256    // GEMM1 N / GEMM2 K
#define COUT  256    // GEMM2 N
#define MROWS 64     // M rows per block (4 sub-tiles of 16)

typedef __attribute__((ext_vector_type(16))) __bf16        v16bf;
typedef __attribute__((ext_vector_type(8)))  float         v8f;
typedef __attribute__((ext_vector_type(4)))  unsigned int  v4u;

union Frag16 { v4u q[2]; v16bf bf; };

static __device__ __forceinline__ unsigned short f2bf(float f) {
    unsigned int u = __float_as_uint(f);
    u += 0x7fffu + ((u >> 16) & 1u);         // round-to-nearest-even
    return (unsigned short)(u >> 16);
}

// ---------------------------------------------------------------------------
// Kernel 1: 3-NN search (fp32 VALU). 256 blocks x 256 threads = 65536 fine pts.
// Coarse positions for the block's batch staged in LDS (12 KB).
// ---------------------------------------------------------------------------
__global__ void knn_kernel(const float* __restrict__ pos,
                           const float* __restrict__ pos_skip,
                           float* __restrict__ w_out,
                           int* __restrict__ i_out) {
    __shared__ float sp[NC * 3];
    const int t = threadIdx.x;
    const int b = blockIdx.x >> 4;           // 16 blocks per batch (4096/256)
    const float* pc = pos + (size_t)b * NC * 3;
    for (int i = t; i < NC * 3; i += 256) sp[i] = pc[i];
    __syncthreads();

    const int f = blockIdx.x * 256 + t;      // global fine-point row
    const float px = pos_skip[f * 3 + 0];
    const float py = pos_skip[f * 3 + 1];
    const float pz = pos_skip[f * 3 + 2];

    float d0 = 1e30f, d1 = 1e30f, d2 = 1e30f;
    int   i0 = 0, i1 = 0, i2 = 0;
    for (int j = 0; j < NC; ++j) {
        const float dx = px - sp[j * 3 + 0];
        const float dy = py - sp[j * 3 + 1];
        const float dz = pz - sp[j * 3 + 2];
        const float d = dx * dx + dy * dy + dz * dz;
        if (d < d2) {
            if (d < d1) {
                d2 = d1; i2 = i1;
                if (d < d0) { d1 = d0; i1 = i0; d0 = d; i0 = j; }
                else        { d1 = d;  i1 = j; }
            } else { d2 = d; i2 = j; }
        }
    }
    const float w0 = 1.0f / fmaxf(d0, 1e-16f);
    const float w1 = 1.0f / fmaxf(d1, 1e-16f);
    const float w2 = 1.0f / fmaxf(d2, 1e-16f);
    const float s  = 1.0f / (w0 + w1 + w2);
    w_out[f * 3 + 0] = w0 * s;
    w_out[f * 3 + 1] = w1 * s;
    w_out[f * 3 + 2] = w2 * s;
    const int base = b * NC;
    i_out[f * 3 + 0] = base + i0;
    i_out[f * 3 + 1] = base + i1;
    i_out[f * 3 + 2] = base + i2;
}

// ---------------------------------------------------------------------------
// Kernel 2: convert W1/W2 (f32, KxN row-major) into bf16 WMMA B-fragment
// order. Fragment dword for (ktile, ntile, lane, v):
//   half = lane>>4, ncol = lane&15, n = ntile*16+ncol, k0 = ktile*32+16*half+2v
//   dword = pack_bf16( W[k0][n], W[k0+1][n] )
// Flat output index = ((ktile*16 + ntile)*32 + lane)*8 + v.
// W1: 12 ktiles -> 49152 dwords. W2: 8 ktiles -> 32768 dwords. Grid covers both.
// ---------------------------------------------------------------------------
__global__ void prep_kernel(const float* __restrict__ W1,
                            const float* __restrict__ W2,
                            unsigned int* __restrict__ wf1,
                            unsigned int* __restrict__ wf2) {
    int t = blockIdx.x * 256 + threadIdx.x;   // 0 .. 81919
    const float* W;
    unsigned int* out;
    if (t < 12 * 16 * 32 * 8) { W = W1; out = wf1; }
    else                      { W = W2; out = wf2; t -= 12 * 16 * 32 * 8; }
    const int v    = t & 7;
    const int lane = (t >> 3) & 31;
    const int tile = t >> 8;
    const int nt   = tile & 15;
    const int kt   = tile >> 4;
    const int half = lane >> 4;
    const int ncol = lane & 15;
    const int n    = nt * 16 + ncol;
    const int k0   = kt * 32 + half * 16 + v * 2;
    const unsigned int lo = f2bf(W[(size_t)k0 * 256 + n]);
    const unsigned int hi = f2bf(W[(size_t)(k0 + 1) * 256 + n]);
    out[t] = lo | (hi << 16);
}

// ---------------------------------------------------------------------------
// Kernel 3: fused interpolate + concat + GEMM1(relu) + GEMM2(relu).
// 1024 blocks x 256 threads (8 waves). Block owns 64 rows (4 M-subtiles),
// full N=256; wave w keeps BOTH its N-tiles' accumulators live (2x4x8 VGPRs)
// and loads the 4 A-fragments once per k-step, feeding 8 WMMAs per
// (8 ds_load_b128 + 4 global_load_b128) cluster.
// ---------------------------------------------------------------------------
__global__ void fp_fused_kernel(const float* __restrict__ x,
                                const float* __restrict__ x_skip,
                                const float* __restrict__ b1,
                                const float* __restrict__ b2,
                                const float* __restrict__ knnW,
                                const int*   __restrict__ knnI,
                                const unsigned int* __restrict__ wf1,
                                const unsigned int* __restrict__ wf2,
                                float* __restrict__ out) {
    __shared__ __align__(16) unsigned short hA[MROWS * CH1];   // 48 KB bf16 [64][384]
    __shared__ __align__(16) unsigned short hB[MROWS * CMID];  // 32 KB bf16 [64][256]

    const int t = threadIdx.x;
    const int rowbase = blockIdx.x * MROWS;

    // ---- stage h = [knn-interp(x) | x_skip] as bf16 in LDS (64 rows) ----
    {
        const int cl = t & 15;          // column lane (16 per row -> coalesced 64B)
        #pragma unroll
        for (int g = 0; g < 4; ++g) {
            const int r = g * 16 + (t >> 4);
            const int f = rowbase + r;
            const float w0 = knnW[f * 3 + 0];
            const float w1 = knnW[f * 3 + 1];
            const float w2 = knnW[f * 3 + 2];
            const float* p0 = x + (size_t)knnI[f * 3 + 0] * CIN;
            const float* p1 = x + (size_t)knnI[f * 3 + 1] * CIN;
            const float* p2 = x + (size_t)knnI[f * 3 + 2] * CIN;
            #pragma unroll
            for (int c = cl; c < CIN; c += 16) {
                const float v = w0 * p0[c] + w1 * p1[c] + w2 * p2[c];
                hA[r * CH1 + c] = f2bf(v);
            }
            const float* ps = x_skip + (size_t)f * CSK;
            #pragma unroll
            for (int c = cl; c < CSK; c += 16) {
                hA[r * CH1 + CIN + c] = f2bf(ps[c]);
            }
        }
    }
    __syncthreads();

    const int wave = t >> 5;
    const int lane = t & 31;
    const int half = lane >> 4;
    const int ncol = lane & 15;
    const int arow = lane & 15;          // A-fragment row within 16-row subtile

    // ---- GEMM1: h[64x384] @ Wf1 + b1, relu -> hB (bf16) ----
    {
        v8f acc[2][4];
        #pragma unroll
        for (int s = 0; s < 2; ++s) {
            const float bias = b1[(wave * 2 + s) * 16 + ncol];
            #pragma unroll
            for (int m = 0; m < 4; ++m)
                #pragma unroll
                for (int r = 0; r < 8; ++r) acc[s][m][r] = bias;
        }

        #pragma unroll
        for (int kt = 0; kt < 12; ++kt) {
            Frag16 a[4];
            #pragma unroll
            for (int m = 0; m < 4; ++m) {
                const unsigned short* ap =
                    &hA[(m * 16 + arow) * CH1 + kt * 32 + half * 16];
                a[m].q[0] = *(const v4u*)(ap);       // VGPR 0..3 : K = 16*half + 0..7
                a[m].q[1] = *(const v4u*)(ap + 8);   // VGPR 4..7 : K = 16*half + 8..15
            }
            #pragma unroll
            for (int s = 0; s < 2; ++s) {
                const int nt = wave * 2 + s;
                Frag16 bfr;
                const unsigned int* bp =
                    wf1 + (((size_t)kt * 16 + nt) * 32 + lane) * 8;
                bfr.q[0] = *(const v4u*)(bp);
                bfr.q[1] = *(const v4u*)(bp + 4);
                #pragma unroll
                for (int m = 0; m < 4; ++m)
                    acc[s][m] = __builtin_amdgcn_wmma_f32_16x16x32_bf16(
                        false, a[m].bf, false, bfr.bf, (short)0, acc[s][m],
                        false, false);
            }
        }

        #pragma unroll
        for (int s = 0; s < 2; ++s) {
            const int nt = wave * 2 + s;
            #pragma unroll
            for (int m = 0; m < 4; ++m)
                #pragma unroll
                for (int r = 0; r < 8; ++r) {
                    const float v = fmaxf(acc[s][m][r], 0.0f);
                    hB[(m * 16 + half * 8 + r) * CMID + nt * 16 + ncol] = f2bf(v);
                }
        }
    }
    __syncthreads();

    // ---- GEMM2: hB[64x256] @ Wf2 + b2, relu -> out (f32) ----
    {
        v8f acc[2][4];
        #pragma unroll
        for (int s = 0; s < 2; ++s) {
            const float bias = b2[(wave * 2 + s) * 16 + ncol];
            #pragma unroll
            for (int m = 0; m < 4; ++m)
                #pragma unroll
                for (int r = 0; r < 8; ++r) acc[s][m][r] = bias;
        }

        #pragma unroll
        for (int kt = 0; kt < 8; ++kt) {
            Frag16 a[4];
            #pragma unroll
            for (int m = 0; m < 4; ++m) {
                const unsigned short* ap =
                    &hB[(m * 16 + arow) * CMID + kt * 32 + half * 16];
                a[m].q[0] = *(const v4u*)(ap);
                a[m].q[1] = *(const v4u*)(ap + 8);
            }
            #pragma unroll
            for (int s = 0; s < 2; ++s) {
                const int nt = wave * 2 + s;
                Frag16 bfr;
                const unsigned int* bp =
                    wf2 + (((size_t)kt * 16 + nt) * 32 + lane) * 8;
                bfr.q[0] = *(const v4u*)(bp);
                bfr.q[1] = *(const v4u*)(bp + 4);
                #pragma unroll
                for (int m = 0; m < 4; ++m)
                    acc[s][m] = __builtin_amdgcn_wmma_f32_16x16x32_bf16(
                        false, a[m].bf, false, bfr.bf, (short)0, acc[s][m],
                        false, false);
            }
        }

        #pragma unroll
        for (int s = 0; s < 2; ++s) {
            const int nt = wave * 2 + s;
            #pragma unroll
            for (int m = 0; m < 4; ++m)
                #pragma unroll
                for (int r = 0; r < 8; ++r) {
                    const int grow = rowbase + m * 16 + half * 8 + r;
                    out[(size_t)grow * COUT + nt * 16 + ncol] =
                        fmaxf(acc[s][m][r], 0.0f);
                }
        }
    }
}

// ---------------------------------------------------------------------------
extern "C" void kernel_launch(void* const* d_in, const int* in_sizes, int n_in,
                              void* d_out, int out_size, void* d_ws, size_t ws_size,
                              hipStream_t stream) {
    const float* x        = (const float*)d_in[0];   // [16384, 256]
    const float* pos      = (const float*)d_in[1];   // [16384, 3]
    const float* x_skip   = (const float*)d_in[3];   // [65536, 128]
    const float* pos_skip = (const float*)d_in[4];   // [65536, 3]
    const float* W1       = (const float*)d_in[6];   // [384, 256]
    const float* b1       = (const float*)d_in[7];   // [256]
    const float* W2       = (const float*)d_in[8];   // [256, 256]
    const float* b2       = (const float*)d_in[9];   // [256]
    float*       out      = (float*)d_out;           // [65536, 256]

    char* ws = (char*)d_ws;
    float*        knnW = (float*)(ws);                          // 786432 B
    int*          knnI = (int*)(ws + 786432);                   // 786432 B
    unsigned int* wf1  = (unsigned int*)(ws + 1572864);         // 196608 B
    unsigned int* wf2  = (unsigned int*)(ws + 1572864 + 196608);// 131072 B

    knn_kernel<<<256, 256, 0, stream>>>(pos, pos_skip, knnW, knnI);
    prep_kernel<<<320, 256, 0, stream>>>(W1, W2, wf1, wf2);
    fp_fused_kernel<<<1024, 256, 0, stream>>>(x, x_skip, b1, b2,
                                              knnW, knnI, wf1, wf2, out);
}
